// MAGE_Block_28527172780645
// MI455X (gfx1250) — matrix operand order repaired and checked by reading, not compiled, not run on hardware
//
#include <hip/hip_runtime.h>

// ---------------------------------------------------------------------------
// Problem constants (match reference)
// ---------------------------------------------------------------------------
#define BB   64
#define NN   512
#define FF   64
#define KK   16
#define GG   32
#define MDD  64
#define TOPK 4
#define EE   2
#define C2   (EE * KK * GG)   // 1024 : stage-2 contraction length
#define EK   (EE * KK)        // 32   : fused (e,k) index

typedef __attribute__((ext_vector_type(16))) __bf16    v16bf;
typedef __attribute__((ext_vector_type(8)))  __bf16    v8bf;
typedef __attribute__((ext_vector_type(8)))  float     v8f;
typedef __attribute__((ext_vector_type(8)))  unsigned  v8u;

static __device__ __forceinline__ v16bf cat8(v8bf lo, v8bf hi) {
    return __builtin_shufflevector(lo, hi, 0, 1, 2, 3, 4, 5, 6, 7,
                                   8, 9, 10, 11, 12, 13, 14, 15);
}

// ---------------------------------------------------------------------------
// 1) coef = [1, -(exp(<q1,k1>) - exp(<q2,k2>) + lambda_init)]
// ---------------------------------------------------------------------------
__global__ void coef_kernel(const float* __restrict__ lq1, const float* __restrict__ lk1,
                            const float* __restrict__ lq2, const float* __restrict__ lk2,
                            float* __restrict__ coef) {
    __shared__ float s1[MDD], s2[MDD];
    int t = threadIdx.x;
    s1[t] = lq1[t] * lk1[t];
    s2[t] = lq2[t] * lk2[t];
    __syncthreads();
    for (int s = MDD / 2; s > 0; s >>= 1) {
        if (t < s) { s1[t] += s1[t + s]; s2[t] += s2[t + s]; }
        __syncthreads();
    }
    if (t == 0) {
        float lambda_init = 0.8f - 0.6f * __expf(-0.3f);   // DEPTH = 1
        coef[0] = 1.0f;
        coef[1] = -(__expf(s1[0]) - __expf(s2[0]) + lambda_init);
    }
}

// ---------------------------------------------------------------------------
// 2) Router GEMM + sigmoid + top-4 gate. Writes coef-scaled route
//    routec[b,n,e*16+k] = coef[e] * route[b,n,k]  (f32, ws) and
//    topk_idx (int32, second output region of d_out).
// ---------------------------------------------------------------------------
__global__ void router_kernel(const float* __restrict__ x,
                              const float* __restrict__ rw,
                              const float* __restrict__ rb,
                              const int*   __restrict__ biasp,
                              const float* __restrict__ coef,
                              float* __restrict__ routec,
                              int*   __restrict__ topk) {
    __shared__ float w[KK * FF];
    for (int i = threadIdx.x; i < KK * FF; i += 256) w[i] = rw[i];
    __syncthreads();

    int tkn = blockIdx.x * 256 + threadIdx.x;       // 0 .. B*N-1
    const float* xr = x + tkn * FF;
    float xv[FF];
#pragma unroll
    for (int i = 0; i < FF / 4; ++i) {
        float4 v = reinterpret_cast<const float4*>(xr)[i];
        xv[4 * i + 0] = v.x; xv[4 * i + 1] = v.y;
        xv[4 * i + 2] = v.z; xv[4 * i + 3] = v.w;
    }

    float biasf = (float)biasp[0];
    float sig[KK], sel[KK];
#pragma unroll
    for (int k = 0; k < KK; ++k) {
        float z = rb[k];
#pragma unroll
        for (int f = 0; f < FF; ++f) z += xv[f] * w[k * FF + f];
        sig[k] = 1.0f / (1.0f + __expf(-z));
        sel[k] = sig[k] + biasf;
    }

    unsigned m = 0;
    float sum = 0.0f;
    int idx4[TOPK];
#pragma unroll
    for (int t = 0; t < TOPK; ++t) {
        int bi = 0; float bv = -3.0e38f;
#pragma unroll
        for (int k = 0; k < KK; ++k)
            if (!((m >> k) & 1u) && sel[k] > bv) { bv = sel[k]; bi = k; }
        m |= 1u << bi;
        idx4[t] = bi;
        sum += sig[bi];
    }
    float inv = 1.0f / sum;
    float c0 = coef[0], c1 = coef[1];
#pragma unroll
    for (int k = 0; k < KK; ++k) {
        float r = ((m >> k) & 1u) ? sig[k] * inv : 0.0f;
        routec[tkn * EK + k]      = c0 * r;
        routec[tkn * EK + KK + k] = c1 * r;
    }
#pragma unroll
    for (int t = 0; t < TOPK; ++t) topk[tkn * TOPK + t] = idx4[t];
}

// ---------------------------------------------------------------------------
// 3) softmax(E1, axis=-1) -> bf16, layout E1sb[i][e*512 + k*32 + d]
//    one wave (32 lanes) per row of G=32.
// ---------------------------------------------------------------------------
__global__ void softmax_e1_kernel(const float* __restrict__ E1,
                                  __bf16* __restrict__ E1sb) {
    int wave = threadIdx.x >> 5;
    int lane = threadIdx.x & 31;
    int row  = blockIdx.x * 8 + wave;          // (e*K+k)*N + i, 0..16383
    float v  = E1[row * GG + lane];
    float mx = v;
#pragma unroll
    for (int off = 16; off > 0; off >>= 1) mx = fmaxf(mx, __shfl_xor(mx, off, 32));
    float ex = __expf(v - mx);
    float s  = ex;
#pragma unroll
    for (int off = 16; off > 0; off >>= 1) s += __shfl_xor(s, off, 32);
    int i  = row & (NN - 1);
    int ek = row >> 9;                          // e*K + k
    E1sb[i * C2 + ek * GG + lane] = (__bf16)(ex / s);
}

// ---------------------------------------------------------------------------
// 4) softmax(E2, axis=-1) -> bf16, same layout as input: E2sb[(e,k,d)][j]
//    one block per row of N=512.
// ---------------------------------------------------------------------------
__global__ void softmax_e2_kernel(const float* __restrict__ E2,
                                  __bf16* __restrict__ E2sb) {
    __shared__ float red[256];
    int row = blockIdx.x, t = threadIdx.x;
    float a = E2[row * NN + t];
    float b = E2[row * NN + 256 + t];
    red[t] = fmaxf(a, b);
    __syncthreads();
    for (int s = 128; s > 0; s >>= 1) {
        if (t < s) red[t] = fmaxf(red[t], red[t + s]);
        __syncthreads();
    }
    float mx = red[0];
    __syncthreads();
    float e0 = __expf(a - mx), e1 = __expf(b - mx);
    red[t] = e0 + e1;
    __syncthreads();
    for (int s = 128; s > 0; s >>= 1) {
        if (t < s) red[t] += red[t + s];
        __syncthreads();
    }
    float inv = 1.0f / red[0];
    E2sb[row * NN + t]       = (__bf16)(e0 * inv);
    E2sb[row * NN + 256 + t] = (__bf16)(e1 * inv);
}

// ---------------------------------------------------------------------------
// 5) xT[b][f][j] = bf16(x[b][j][f])  (LDS tile transpose, 64x64 per block)
// ---------------------------------------------------------------------------
__global__ void transpose_x_kernel(const float* __restrict__ x,
                                   __bf16* __restrict__ xT) {
    __shared__ __bf16 tile[64 * 65];
    int b  = blockIdx.x >> 3;
    int j0 = (blockIdx.x & 7) * 64;
    for (int idx = threadIdx.x; idx < 64 * 64; idx += 256) {
        int r = idx >> 6, f = idx & 63;
        tile[f * 65 + r] = (__bf16)x[(b * NN + j0 + r) * FF + f];
    }
    __syncthreads();
    for (int idx = threadIdx.x; idx < 64 * 64; idx += 256) {
        int f = idx >> 6, c = idx & 63;
        xT[(b * FF + f) * NN + j0 + c] = tile[f * 65 + c];
    }
}

// ---------------------------------------------------------------------------
// 6) Stage-1 WMMA:  Y[e,k,:,:|b] = E2s[e,k] (32x512) . x[b] (512x64)
//    stored transposed: YT[b][f][c], c = ek*32 + d.
//    Block = 8 waves; wave -> one 16x16 tile; 16 K-steps of 32 (bf16).
// ---------------------------------------------------------------------------
__global__ void stage1_wmma_kernel(const __bf16* __restrict__ E2sb,
                                   const __bf16* __restrict__ xT,
                                   __bf16* __restrict__ YT) {
    int b    = blockIdx.x & (BB - 1);
    int ek   = blockIdx.x >> 6;                 // 0..31
    int w    = threadIdx.x >> 5;                // 0..7
    int lane = threadIdx.x & 31;
    int d0   = (w >> 2) * 16;                   // 0 or 16
    int f0   = (w & 3) * 16;                    // 0,16,32,48
    int lr   = lane & 15;
    bool hih = lane >= 16;

    const __bf16* arow = E2sb + (ek * GG + d0 + lr) * NN;               // A: row d, cols j
    const __bf16* brow = xT + (b * FF + f0 + lr) * NN + (hih ? 16 : 0); // B: col f, rows j
    int alo = hih ? 16 : 0;
    int ahi = hih ? 24 : 8;

    v8f acc = {};
#pragma unroll
    for (int kk = 0; kk < 16; ++kk) {
        int j0 = kk * 32;
        v16bf a = cat8(*reinterpret_cast<const v8bf*>(arow + j0 + alo),
                       *reinterpret_cast<const v8bf*>(arow + j0 + ahi));
        v16bf bfr = cat8(*reinterpret_cast<const v8bf*>(brow + j0),
                         *reinterpret_cast<const v8bf*>(brow + j0 + 8));
        acc = __builtin_amdgcn_wmma_f32_16x16x32_bf16(false, a, false, bfr,
                                                      (short)0, acc, false, false);
    }

    // C tile (m,n): lane holds n = lr, m = v + (hih?8:0); store transposed:
    // YT[b][f0+n][ek*32 + d0 + m]  -> 8 contiguous bf16 per lane = one b128.
    int m0 = hih ? 8 : 0;
    __bf16* dst = YT + (b * FF + f0 + lr) * C2 + ek * GG + d0 + m0;
    v8bf o;
#pragma unroll
    for (int v = 0; v < 8; ++v) o[v] = (__bf16)acc[v];
    *reinterpret_cast<v8bf*>(dst) = o;
}

// ---------------------------------------------------------------------------
// 7) Stage-2 WMMA:  out[b] (512x64) = A_b (512x1024) . Y_b (1024x64)
//    A_b[i,c] = routec[b,i,e*16+k]*E1s[e,k,i,d]; scale applied to the bf16
//    A fragment with packed bf16 multiplies (V_PK_MUL_BF16 path).
// ---------------------------------------------------------------------------
__global__ void stage2_wmma_kernel(const __bf16* __restrict__ E1sb,
                                   const __bf16* __restrict__ YT,
                                   const float* __restrict__ routec,
                                   float* __restrict__ out) {
    int b    = blockIdx.x >> 4;
    int ig   = blockIdx.x & 15;
    int w    = threadIdx.x >> 5;
    int lane = threadIdx.x & 31;
    int i0   = ig * 32 + (w >> 2) * 16;
    int f0   = (w & 3) * 16;
    int lr   = lane & 15;
    bool hih = lane >= 16;

    int irow = i0 + lr;
    const __bf16* arow = E1sb + irow * C2;
    const __bf16* brow = YT + (b * FF + f0 + lr) * C2 + (hih ? 16 : 0);
    int alo = hih ? 16 : 0;
    int ahi = hih ? 24 : 8;

    // coef[e]*route[b,i,k] per chunk index cc = e*16+k, pre-packed bf16x2
    unsigned spk[EK];
#pragma unroll
    for (int c = 0; c < EK; ++c) {
        float s = routec[(b * NN + irow) * EK + c];
        unsigned sb = (unsigned)__builtin_bit_cast(unsigned short, (__bf16)s);
        spk[c] = sb * 0x10001u;
    }

    v8f acc = {};
#pragma unroll 8
    for (int cc = 0; cc < 32; ++cc) {
        int c0 = cc * 32;                       // one (e,k) per 32-chunk
        v16bf a = cat8(*reinterpret_cast<const v8bf*>(arow + c0 + alo),
                       *reinterpret_cast<const v8bf*>(arow + c0 + ahi));
        v8u sp;
#pragma unroll
        for (int t = 0; t < 8; ++t) sp[t] = spk[cc];
        a = a * __builtin_bit_cast(v16bf, sp);  // packed bf16 scale
        v16bf bfr = cat8(*reinterpret_cast<const v8bf*>(brow + c0),
                         *reinterpret_cast<const v8bf*>(brow + c0 + 8));
        acc = __builtin_amdgcn_wmma_f32_16x16x32_bf16(false, a, false, bfr,
                                                      (short)0, acc, false, false);
    }

    int m0 = hih ? 8 : 0;
    float* drow = out + (b * NN + i0 + m0) * FF + f0 + lr;
#pragma unroll
    for (int v = 0; v < 8; ++v) drow[v * FF] = acc[v];
}

// ---------------------------------------------------------------------------
// Launch
// ---------------------------------------------------------------------------
extern "C" void kernel_launch(void* const* d_in, const int* in_sizes, int n_in,
                              void* d_out, int out_size, void* d_ws, size_t ws_size,
                              hipStream_t stream) {
    (void)in_sizes; (void)n_in; (void)out_size; (void)ws_size;

    const float* x        = (const float*)d_in[0];
    const float* E1       = (const float*)d_in[1];
    const float* E2       = (const float*)d_in[2];
    const float* router_w = (const float*)d_in[3];
    const float* router_b = (const float*)d_in[4];
    // d_in[5] = outer_lambda (unused: outer == identity)
    const float* lq1      = (const float*)d_in[6];
    const float* lk1      = (const float*)d_in[7];
    const float* lq2      = (const float*)d_in[8];
    const float* lk2      = (const float*)d_in[9];
    const int*   biasp    = (const int*)d_in[10];

    float* out  = (float*)d_out;
    int*   topk = (int*)d_out + (size_t)BB * NN * FF;

    // workspace carve-up (byte offsets, all >=16B aligned)
    char* ws = (char*)d_ws;
    float*  routec = (float*)(ws + 0);                        // 4 MB
    float*  coef   = (float*)(ws + 4194304);                  // 8 B (pad 1 KB)
    __bf16* E1sb   = (__bf16*)(ws + 4195328);                 // 1 MB
    __bf16* E2sb   = (__bf16*)(ws + 5243904);                 // 1 MB
    __bf16* xT     = (__bf16*)(ws + 6292480);                 // 4 MB
    __bf16* YT     = (__bf16*)(ws + 10486784);                // 8 MB

    coef_kernel<<<1, MDD, 0, stream>>>(lq1, lk1, lq2, lk2, coef);
    router_kernel<<<(BB * NN) / 256, 256, 0, stream>>>(x, router_w, router_b,
                                                       biasp, coef, routec, topk);
    softmax_e1_kernel<<<(EE * KK * NN) / 8, 256, 0, stream>>>(E1, E1sb);
    softmax_e2_kernel<<<EE * KK * GG, 256, 0, stream>>>(E2, E2sb);
    transpose_x_kernel<<<BB * (NN / 64), 256, 0, stream>>>(x, xT);
    stage1_wmma_kernel<<<EE * KK * BB, 256, 0, stream>>>(E2sb, xT, YT);
    stage2_wmma_kernel<<<BB * (NN / 32), 256, 0, stream>>>(E1sb, YT, routec, out);
}
